// Attn_6528350289920
// MI455X (gfx1250) — matrix operand order repaired and checked by reading, not compiled, not run on hardware
//
#include <hip/hip_runtime.h>

typedef _Float16 half_t;
typedef __attribute__((ext_vector_type(16))) _Float16 v16h;
typedef __attribute__((ext_vector_type(8)))  _Float16 v8h;
typedef __attribute__((ext_vector_type(8)))  float    v8f;

#define B_SZ     4
#define NSELF    2048
#define NCROSS   256
#define S_TOT    2304      // NSELF + NCROSS
#define DM       1024
#define NH       16
#define HD       64
#define QK_SCALE 0.03125f  // 1024^-0.5
#define ATTN_SCALE 8.0f    // 64^0.5  (folded into Q)
#define LN10000  9.210340371976184f

// ---- CDNA5 async global->LDS (ASYNCcnt-tracked DMA), per-lane 16B ----------
__device__ __forceinline__ void async_ld_b128(void* lds, const void* gmem) {
  // flat LDS pointer low 32 bits == wave-relative LDS byte offset (aperture rule)
  unsigned l = (unsigned)(unsigned long long)lds;
  asm volatile("global_load_async_to_lds_b128 %0, %1, off"
               :: "v"(l), "v"(gmem) : "memory");
}
__device__ __forceinline__ void wait_async0() {
  asm volatile("s_wait_asynccnt 0x0" ::: "memory");
}

// ---------------------------------------------------------------- convert
__global__ void f32_to_f16(const float* __restrict__ in, half_t* __restrict__ out, size_t n) {
  size_t i = (size_t)blockIdx.x * blockDim.x + threadIdx.x;
  size_t stride = (size_t)gridDim.x * blockDim.x;
  for (; i < n; i += stride) out[i] = (half_t)in[i];
}

// ---------------------------------------------------------------- GEMM (NT)
// C[m,n] = sum_k A[m,k] * W[n,k]   A: MxK f16, W: NxK f16 (row-major)
// block tile 128x64, BK=32, 8 waves, 2x2 WMMA tiles/wave,
// double-buffered LDS fed by global_load_async_to_lds_b128.
template<bool OUT_F32, bool BIAS>
__global__ __launch_bounds__(256) void gemm_nt(
    const half_t* __restrict__ A, const half_t* __restrict__ W,
    void* __restrict__ Cout, const float* __restrict__ bias,
    int M, int N, int K)
{
  constexpr int BM = 128, BN = 64, BK = 32;
  constexpr int LDA = BK + 8, LDB = BK + 8;  // 16B-aligned padded rows
  __shared__ __align__(16) half_t As[2][BM * LDA];
  __shared__ __align__(16) half_t Bs[2][BN * LDB];

  const int tid  = threadIdx.x;
  const int lane = tid & 31;
  const int wave = tid >> 5;     // 0..7
  const int wm   = wave & 3;     // m offset 32*wm
  const int wn   = wave >> 2;    // n offset 32*wn
  const int l16  = lane & 15;
  const int lhi  = lane >> 4;    // 0/1
  const int bm0  = blockIdx.x * BM;
  const int bn0  = blockIdx.y * BN;

  // staging pattern: A = 512 16B-chunks (2/thread), B = 256 chunks (1/thread)
  const int ra = tid >> 2, ca = (tid & 3) * 8;   // A rows ra, ra+64
  const int rb = tid >> 2, cb = (tid & 3) * 8;   // B row rb

  auto stage = [&](int buf, int k0) {
    async_ld_b128(&As[buf][ra * LDA + ca],        &A[(size_t)(bm0 + ra) * K + k0 + ca]);
    async_ld_b128(&As[buf][(ra + 64) * LDA + ca], &A[(size_t)(bm0 + ra + 64) * K + k0 + ca]);
    async_ld_b128(&Bs[buf][rb * LDB + cb],        &W[(size_t)(bn0 + rb) * K + k0 + cb]);
  };

  v8f acc[2][2] = {};

  stage(0, 0);
  wait_async0();
  __syncthreads();

  int buf = 0;
  for (int k0 = 0; k0 < K; k0 += BK) {
    if (k0 + BK < K) stage(buf ^ 1, k0 + BK);

    v16h afrag[2], bfrag[2];
    #pragma unroll
    for (int tm = 0; tm < 2; ++tm) {
      const half_t* p = &As[buf][(wm * 32 + tm * 16 + l16) * LDA];
      v8h lo = *(const v8h*)&p[lhi ? 16 : 0];   // K 0-7   | 16-23
      v8h hi = *(const v8h*)&p[lhi ? 24 : 8];   // K 8-15  | 24-31
      #pragma unroll
      for (int i = 0; i < 8; ++i) { afrag[tm][i] = lo[i]; afrag[tm][8 + i] = hi[i]; }
    }
    #pragma unroll
    for (int tn = 0; tn < 2; ++tn) {
      const half_t* p = &Bs[buf][(wn * 32 + tn * 16 + l16) * LDB + (lhi ? 16 : 0)];
      v8h lo = *(const v8h*)&p[0];
      v8h hi = *(const v8h*)&p[8];
      #pragma unroll
      for (int i = 0; i < 8; ++i) { bfrag[tn][i] = lo[i]; bfrag[tn][8 + i] = hi[i]; }
    }
    #pragma unroll
    for (int tm = 0; tm < 2; ++tm)
      #pragma unroll
      for (int tn = 0; tn < 2; ++tn)
        acc[tm][tn] = __builtin_amdgcn_wmma_f32_16x16x32_f16(
            false, afrag[tm], false, bfrag[tn], (short)0, acc[tm][tn], false, false);

    wait_async0();      // next tile landed in LDS
    __syncthreads();    // all waves done with current tile
    buf ^= 1;
  }

  // epilogue: D layout -> VGPR j holds (M=j | M=j+8), N = l16
  #pragma unroll
  for (int tm = 0; tm < 2; ++tm)
    #pragma unroll
    for (int tn = 0; tn < 2; ++tn) {
      int m0 = bm0 + wm * 32 + tm * 16 + (lhi ? 8 : 0);
      int n  = bn0 + wn * 32 + tn * 16 + l16;
      float bv = BIAS ? bias[n] : 0.f;
      #pragma unroll
      for (int j = 0; j < 8; ++j) {
        float v = acc[tm][tn][j] + bv;
        size_t idx = (size_t)(m0 + j) * N + n;
        if (OUT_F32) ((float*)Cout)[idx] = v;
        else         ((half_t*)Cout)[idx] = (half_t)v;
      }
    }
}

// ---------------------------------------------------------------- QKV post
__global__ __launch_bounds__(256) void qkv_post(
    const half_t* __restrict__ qkvs,   // [B*2048, 3072]
    const half_t* __restrict__ qkvc,   // [B*256, 3072]
    const float* __restrict__ g_self, const float* __restrict__ g_cross,
    half_t* __restrict__ Qh, half_t* __restrict__ Kh, half_t* __restrict__ Vt)
{
  int gid = blockIdx.x * blockDim.x + threadIdx.x;
  if (gid >= B_SZ * NH * S_TOT) return;
  int pos = gid % S_TOT;
  int h   = (gid / S_TOT) % NH;
  int b   = gid / (S_TOT * NH);
  bool self = pos < NSELF;
  const half_t* src = self ? &qkvs[((size_t)b * NSELF + pos) * (3 * DM) + h * HD]
                           : &qkvc[((size_t)b * NCROSS + (pos - NSELF)) * (3 * DM) + h * HD];
  const float* g = (self ? g_self : g_cross) + h * HD;

  float q[HD], k[HD];
  float qs = 0.f, ks = 0.f;
  #pragma unroll
  for (int i = 0; i < HD; ++i) {
    q[i] = (float)src[i];           qs += q[i] * q[i];
    k[i] = (float)src[DM + i];      ks += k[i] * k[i];
  }
  float qn = QK_SCALE * ATTN_SCALE / fmaxf(sqrtf(qs), 1e-12f);
  float kn = QK_SCALE / fmaxf(sqrtf(ks), 1e-12f);

  half_t* qd = Qh + ((size_t)(b * NH + h) * S_TOT + pos) * HD;
  half_t* kd = Kh + ((size_t)(b * NH + h) * S_TOT + pos) * HD;
  float p = (float)pos;
  #pragma unroll
  for (int j = 0; j < HD / 2; ++j) {
    float inv = __expf(-((float)(2 * j) / (float)HD) * LN10000);
    float sn, cs;
    __sincosf(p * inv, &sn, &cs);
    float q1 = q[j] * g[j] * qn, q2 = q[j + 32] * g[j + 32] * qn;
    float k1 = k[j] * g[j] * kn, k2 = k[j + 32] * g[j + 32] * kn;
    qd[j]      = (half_t)(q1 * cs - q2 * sn);
    qd[j + 32] = (half_t)(q2 * cs + q1 * sn);
    kd[j]      = (half_t)(k1 * cs - k2 * sn);
    kd[j + 32] = (half_t)(k2 * cs + k1 * sn);
  }
  half_t* vd = Vt + (size_t)(b * NH + h) * HD * S_TOT;
  #pragma unroll
  for (int i = 0; i < HD; ++i) vd[(size_t)i * S_TOT + pos] = src[2 * DM + i];
}

// ---------------------------------------------------------------- attention
// 4 waves/block, all same (b,h), consecutive 16-query tiles. K/V blocks (32
// keys) staged once per block into double-buffered LDS via async DMA.
// S^T = K·Q^T (softmax over keys is in-lane + one xor-16), O^T = V^T·P.
__global__ __launch_bounds__(128) void attn(
    const half_t* __restrict__ Qh, const half_t* __restrict__ Kh,
    const half_t* __restrict__ Vt, half_t* __restrict__ Oa)
{
  constexpr int KLD = HD + 8;   // 72 halfs, 144B rows (16B multiple)
  constexpr int VLD = 32 + 8;   // 40 halfs, 80B rows
  __shared__ __align__(16) half_t Ks[2][32 * KLD];
  __shared__ __align__(16) half_t Vs[2][64 * VLD];

  const int tid  = threadIdx.x;           // 0..127
  const int wave = blockIdx.x * 4 + (tid >> 5);
  const int lane = tid & 31;
  const int l16  = lane & 15;
  const int lhi  = lane >> 4;
  const int qt   = wave % (NSELF / 16);
  const int bh   = wave / (NSELF / 16);   // identical for all 4 waves in block
  const int q0   = qt * 16;

  const half_t* Qp = Qh + ((size_t)bh * S_TOT + q0) * HD;
  const half_t* Kp = Kh + (size_t)bh * S_TOT * HD;
  const half_t* Vp = Vt + (size_t)bh * HD * S_TOT;

  // staging: K block 32x64 halfs = 256 chunks, V block 64x32 = 256 chunks;
  // 128 threads -> 2+2 async loads per thread per key-block
  const int rk = tid >> 3, ck = (tid & 7) * 8;
  const int rv = tid >> 2, cv = (tid & 3) * 8;
  auto stage = [&](int buf, int kb) {
    async_ld_b128(&Ks[buf][rk * KLD + ck],        Kp + (size_t)(kb + rk) * HD + ck);
    async_ld_b128(&Ks[buf][(rk + 16) * KLD + ck], Kp + (size_t)(kb + rk + 16) * HD + ck);
    async_ld_b128(&Vs[buf][rv * VLD + cv],        Vp + (size_t)rv * S_TOT + kb + cv);
    async_ld_b128(&Vs[buf][(rv + 32) * VLD + cv], Vp + (size_t)(rv + 32) * S_TOT + kb + cv);
  };

  // Q^T B-operand fragments (two d-steps of 32), reused for all key blocks
  v16h qfrag[2];
  #pragma unroll
  for (int ks = 0; ks < 2; ++ks) {
    const half_t* p = Qp + (size_t)l16 * HD + ks * 32 + lhi * 16;
    v8h lo = *(const v8h*)p;
    v8h hi = *(const v8h*)(p + 8);
    #pragma unroll
    for (int i = 0; i < 8; ++i) { qfrag[ks][i] = lo[i]; qfrag[ks][8 + i] = hi[i]; }
  }

  v8f o[4] = {};
  float mrun = -1e30f, lrun = 0.f;

  stage(0, 0);
  wait_async0();
  __syncthreads();

  int buf = 0;
  for (int kb = 0; kb < S_TOT; kb += 32) {
    if (kb + 32 < S_TOT) stage(buf ^ 1, kb + 32);

    v8f sacc[2] = {};
    #pragma unroll
    for (int t = 0; t < 2; ++t) {
      #pragma unroll
      for (int ks = 0; ks < 2; ++ks) {
        const half_t* p = &Ks[buf][(t * 16 + l16) * KLD + ks * 32];
        v8h lo = *(const v8h*)(p + (lhi ? 16 : 0));
        v8h hi = *(const v8h*)(p + (lhi ? 24 : 8));
        v16h af;
        #pragma unroll
        for (int i = 0; i < 8; ++i) { af[i] = lo[i]; af[8 + i] = hi[i]; }
        sacc[t] = __builtin_amdgcn_wmma_f32_16x16x32_f16(
            false, af, false, qfrag[ks], (short)0, sacc[t], false, false);
      }
    }
    // online softmax over keys (per query column = lane pair l16/l16+16)
    float bm = -1e30f;
    #pragma unroll
    for (int j = 0; j < 8; ++j) bm = fmaxf(bm, fmaxf(sacc[0][j], sacc[1][j]));
    bm = fmaxf(bm, __shfl_xor(bm, 16));
    float mnew = fmaxf(mrun, bm);
    float corr = __expf(mrun - mnew);
    float e0[8], e1[8];
    float bs = 0.f;
    #pragma unroll
    for (int j = 0; j < 8; ++j) {
      e0[j] = __expf(sacc[0][j] - mnew);
      e1[j] = __expf(sacc[1][j] - mnew);
      bs += e0[j] + e1[j];
    }
    bs += __shfl_xor(bs, 16);
    lrun = lrun * corr + bs;
    mrun = mnew;
    #pragma unroll
    for (int t = 0; t < 4; ++t)
      #pragma unroll
      for (int j = 0; j < 8; ++j) o[t][j] *= corr;

    // rebuild P (K=32 keys x N=16 queries) B-operand from e0/e1 via xor-16
    v16h pf;
    #pragma unroll
    for (int i = 0; i < 8; ++i) {
      float a = e0[i], bv = e1[i];
      float asw = __shfl_xor(a, 16);
      float bsw = __shfl_xor(bv, 16);
      pf[i]     = (half_t)(lhi ? bsw : a);   // keys kb+i   | kb+16+i
      pf[8 + i] = (half_t)(lhi ? bv  : asw); // keys kb+8+i | kb+24+i
    }
    // O^T += V^T · P  (4 d-tiles of 16)
    #pragma unroll
    for (int t = 0; t < 4; ++t) {
      const half_t* p = &Vs[buf][(t * 16 + l16) * VLD];
      v8h lo = *(const v8h*)(p + (lhi ? 16 : 0));
      v8h hi = *(const v8h*)(p + (lhi ? 24 : 8));
      v16h vf;
      #pragma unroll
      for (int i = 0; i < 8; ++i) { vf[i] = lo[i]; vf[8 + i] = hi[i]; }
      o[t] = __builtin_amdgcn_wmma_f32_16x16x32_f16(
          false, vf, false, pf, (short)0, o[t], false, false);
    }

    wait_async0();
    __syncthreads();
    buf ^= 1;
  }

  // normalize + store merged heads: Oa[(b*2048+q)*1024 + h*64 + d]
  float inv = 1.f / lrun;
  int b = bh / NH, h = bh % NH;
  half_t* Op = Oa + ((size_t)(b * NSELF + q0 + l16)) * DM + h * HD + lhi * 8;
  #pragma unroll
  for (int t = 0; t < 4; ++t) {
    v8h outv;
    #pragma unroll
    for (int j = 0; j < 8; ++j) outv[j] = (half_t)(o[t][j] * inv);
    *(v8h*)(Op + t * 16) = outv;
  }
}

// ---------------------------------------------------------------- launch
extern "C" void kernel_launch(void* const* d_in, const int* in_sizes, int n_in,
                              void* d_out, int out_size, void* d_ws, size_t ws_size,
                              hipStream_t stream) {
  const float* x      = (const float*)d_in[0];
  const float* c      = (const float*)d_in[1];
  const float* w_qkv  = (const float*)d_in[2];
  const float* w_cqkv = (const float*)d_in[3];
  const float* g_self = (const float*)d_in[4];
  const float* g_cross= (const float*)d_in[5];
  const float* w_out  = (const float*)d_in[6];
  const float* b_out  = (const float*)d_in[7];
  float* out = (float*)d_out;

  char* ws = (char*)d_ws;
  size_t off = 0;
  auto take = [&](size_t bytes) { char* p = ws + off; off = (off + bytes + 255) & ~(size_t)255; return p; };

  const size_t Ms = (size_t)B_SZ * NSELF;     // 8192
  const size_t Mc = (size_t)B_SZ * NCROSS;    // 1024
  half_t* xh    = (half_t*)take(Ms * DM * 2);
  half_t* ch    = (half_t*)take(Mc * DM * 2);
  half_t* wqh   = (half_t*)take((size_t)3 * DM * DM * 2);
  half_t* wcqh  = (half_t*)take((size_t)3 * DM * DM * 2);
  half_t* woh   = (half_t*)take((size_t)DM * DM * 2);
  half_t* qkvs  = (half_t*)take(Ms * 3 * DM * 2);
  half_t* qkvc  = (half_t*)take(Mc * 3 * DM * 2);
  half_t* Qh    = (half_t*)take((size_t)B_SZ * NH * S_TOT * HD * 2);
  half_t* Kh    = (half_t*)take((size_t)B_SZ * NH * S_TOT * HD * 2);
  half_t* Vt    = (half_t*)take((size_t)B_SZ * NH * S_TOT * HD * 2);
  half_t* Oa    = (half_t*)take(Ms * DM * 2);

  // 1) convert inputs/weights to f16
  f32_to_f16<<<4096, 256, 0, stream>>>(x,      xh,   Ms * DM);
  f32_to_f16<<<1024, 256, 0, stream>>>(c,      ch,   Mc * DM);
  f32_to_f16<<<2048, 256, 0, stream>>>(w_qkv,  wqh,  (size_t)3 * DM * DM);
  f32_to_f16<<<2048, 256, 0, stream>>>(w_cqkv, wcqh, (size_t)3 * DM * DM);
  f32_to_f16<<<1024, 256, 0, stream>>>(w_out,  woh,  (size_t)DM * DM);

  // 2) QKV projections (NT GEMM, f16 out)
  gemm_nt<false, false><<<dim3(Ms / 128, (3 * DM) / 64), 256, 0, stream>>>(
      xh, wqh, qkvs, nullptr, (int)Ms, 3 * DM, DM);
  gemm_nt<false, false><<<dim3(Mc / 128, (3 * DM) / 64), 256, 0, stream>>>(
      ch, wcqh, qkvc, nullptr, (int)Mc, 3 * DM, DM);

  // 3) l2norm/gain/scale/RoPE -> Q,K,V(transposed)
  {
    int total = B_SZ * NH * S_TOT;
    qkv_post<<<(total + 255) / 256, 256, 0, stream>>>(qkvs, qkvc, g_self, g_cross, Qh, Kh, Vt);
  }

  // 4) flash attention (self queries only)
  {
    int waves = B_SZ * NH * (NSELF / 16);   // 8192
    attn<<<waves / 4, 128, 0, stream>>>(Qh, Kh, Vt, Oa);
  }

  // 5) output projection with bias, f32 straight into d_out
  gemm_nt<true, true><<<dim3(Ms / 128, DM / 64), 256, 0, stream>>>(
      Oa, woh, out, b_out, (int)Ms, DM, DM);
}
